// Mamba_13726715478163
// MI455X (gfx1250) — matrix operand order repaired and checked by reading, not compiled, not run on hardware
//
#include <hip/hip_runtime.h>
#include <hip/hip_bf16.h>

// ---------------------------------------------------------------------------
// Mamba forward for MI455X (gfx1250).  All GEMMs use v_wmma_f32_16x16x32_bf16
// (one wave -> 16x64 output strip, A-fragment reused across 4 B-fragments).
// Selective scan is chunk-parallelized (associative recurrence): pass1 builds
// per-chunk (end-state, decay-product) summaries, a tiny fix-up kernel scans
// the summaries, pass2 re-runs each chunk seeded with its true initial state.
// ---------------------------------------------------------------------------

#define B_SZ   2
#define SEQ    1024
#define DM     768
#define DI     1536      // d_inner
#define E2     3072      // 2*d_inner
#define DS     16        // d_state
#define RK     48        // dt_rank
#define RKP    64        // dt_rank padded to K-multiple of 32
#define NS     6         // B_SZ * 3 directions
#define NCH    8         // scan chunks along SEQ
#define CL     (SEQ / NCH)

typedef __bf16 bf16_t;
typedef __attribute__((ext_vector_type(16))) __bf16 v16bf;
typedef __attribute__((ext_vector_type(8)))  __bf16 v8bf;
typedef __attribute__((ext_vector_type(8)))  float  v8f;

__device__ __forceinline__ bf16_t f2bf(float f) {
    union { unsigned int u; float f; } in; in.f = f;
    unsigned int u = in.u;
    u = u + 0x7FFFu + ((u >> 16) & 1u);          // round-to-nearest-even
    union { unsigned short s; bf16_t b; } out;
    out.s = (unsigned short)(u >> 16);
    return out.b;
}

__device__ __forceinline__ float softplusf(float x) {
    return (x > 20.0f) ? x : __logf(1.0f + __expf(x));
}

// ------------------------------- converters --------------------------------
__global__ __launch_bounds__(256) void cvt_bf16_kernel(
        const float* __restrict__ in, bf16_t* __restrict__ out, int n) {
    int i = blockIdx.x * 256 + threadIdx.x;
    if (i < n) out[i] = f2bf(in[i]);
}

// W_dt (1536 x 48) -> bf16 (1536 x 64) zero-padded
__global__ __launch_bounds__(256) void cvt_wdt_kernel(
        const float* __restrict__ wdt, bf16_t* __restrict__ out) {
    int i = blockIdx.x * 256 + threadIdx.x;
    if (i >= DI * RKP) return;
    int r = i & (RKP - 1), d = i >> 6;
    out[i] = (r < RK) ? f2bf(wdt[d * RK + r]) : f2bf(0.0f);
}

// x_dbl rows 0..47 (s,80,L) -> dt_r^T bf16 (s, L, 64) zero-padded
__global__ __launch_bounds__(256) void cvt_dtr_kernel(
        const float* __restrict__ xdbl, bf16_t* __restrict__ dtrt) {
    int t = blockIdx.x * 256 + threadIdx.x;      // over NS*SEQ
    if (t >= NS * SEQ) return;
    int s = t >> 10, l = t & (SEQ - 1);
    const float* src = xdbl + (size_t)s * 80 * SEQ;
    bf16_t* dst = dtrt + (size_t)t * RKP;
    #pragma unroll 4
    for (int r = 0; r < RK; ++r) dst[r] = f2bf(src[r * SEQ + l]);
    #pragma unroll
    for (int r = RK; r < RKP; ++r) dst[r] = f2bf(0.0f);
}

// ------------------------------ WMMA GEMM ----------------------------------
// C(MxN,f32,row-major) = A(MxK,bf16,row-major) * B,  B given TRANSPOSED as
// BT(NxK,bf16,row-major).  Block = 4 waves; wave w -> rows [by*64+w*16, +16),
// cols [bx*64, +64) as 4 accumulators.  K multiple of 32, N multiple of 64,
// M multiple of 16 (waves past M early-exit).
__global__ __launch_bounds__(128) void gemm_bf16_wmma(
        const bf16_t* __restrict__ A,  long long lda, long long strideA,
        const bf16_t* __restrict__ BT, long long ldb, long long strideB,
        float* __restrict__ C,         long long ldc, long long strideC,
        int M, int N, int K) {
    const int bat = blockIdx.z;
    A  += (size_t)bat * strideA;
    BT += (size_t)bat * strideB;
    C  += (size_t)bat * strideC;

    const int wave = threadIdx.x >> 5;
    const int lane = threadIdx.x & 31;
    const int ln   = lane & 15;
    const int hi   = lane >> 4;                       // 0/1 (lane half)

    const int row0 = blockIdx.y * 64 + wave * 16;
    if (row0 >= M) return;
    const int col0 = blockIdx.x * 64;

    v8f acc0 = {}, acc1 = {}, acc2 = {}, acc3 = {};

    // A-fragment (16x32 bf16): lane holds row (row0+ln); element e -> K =
    // kc + 16*hi + e (contiguous, matching the B-fragment)
    const bf16_t* arow = A + (size_t)(row0 + ln) * lda;
    // B-fragment (32x16 bf16): lane holds column; 16 contiguous K values
    // starting at kc + 16*hi  ->  one 32B load from BT row.
    const bf16_t* b0 = BT + (size_t)(col0 +  0 + ln) * ldb + hi * 16;
    const bf16_t* b1 = BT + (size_t)(col0 + 16 + ln) * ldb + hi * 16;
    const bf16_t* b2 = BT + (size_t)(col0 + 32 + ln) * ldb + hi * 16;
    const bf16_t* b3 = BT + (size_t)(col0 + 48 + ln) * ldb + hi * 16;

    for (int kc = 0; kc < K; kc += 32) {
        union { v16bf v; v8bf h[2]; } a;
        a.h[0] = *(const v8bf*)(arow + kc + hi * 16);
        a.h[1] = *(const v8bf*)(arow + kc + hi * 16 + 8);
        v16bf fb0 = *(const v16bf*)(b0 + kc);
        v16bf fb1 = *(const v16bf*)(b1 + kc);
        v16bf fb2 = *(const v16bf*)(b2 + kc);
        v16bf fb3 = *(const v16bf*)(b3 + kc);
        acc0 = __builtin_amdgcn_wmma_f32_16x16x32_bf16(false, a.v, false, fb0,
                                                       (short)0, acc0, false, false);
        acc1 = __builtin_amdgcn_wmma_f32_16x16x32_bf16(false, a.v, false, fb1,
                                                       (short)0, acc1, false, false);
        acc2 = __builtin_amdgcn_wmma_f32_16x16x32_bf16(false, a.v, false, fb2,
                                                       (short)0, acc2, false, false);
        acc3 = __builtin_amdgcn_wmma_f32_16x16x32_bf16(false, a.v, false, fb3,
                                                       (short)0, acc3, false, false);
    }

    // C/D layout: VGPR r -> (M = row0 + r + 8*hi, N = col0 + j*16 + ln)
    #pragma unroll
    for (int r = 0; r < 8; ++r) {
        float* crow = C + (size_t)(row0 + r + hi * 8) * ldc + col0 + ln;
        crow[ 0] = acc0[r];
        crow[16] = acc1[r];
        crow[32] = acc2[r];
        crow[48] = acc3[r];
    }
}

// ------------------------- conv4 + SiLU + z-prep ---------------------------
// xz (B_SZ, E2, SEQ).  Stream s = 3*b + dir reads column P_s(l) of rows
// [d] (conv input) and [DI+d] (gate z).  Writes (s,l,d)-layout tensors.
__global__ __launch_bounds__(256) void conv_silu_kernel(
        const float* __restrict__ xz, const float* __restrict__ conv_w,
        const float* __restrict__ conv_b,
        const int* __restrict__ order, const int* __restrict__ order_rev,
        bf16_t* __restrict__ xct, float* __restrict__ xcf,
        float* __restrict__ zs) {
    const int s = blockIdx.y;
    const int d = blockIdx.x * 256 + threadIdx.x;
    const int b = s / 3, dir = s % 3;
    const float w0 = conv_w[d * 4 + 0], w1 = conv_w[d * 4 + 1];
    const float w2 = conv_w[d * 4 + 2], w3 = conv_w[d * 4 + 3];
    const float bias = conv_b[d];
    const float* xrow = xz + ((size_t)b * E2 + d) * SEQ;
    const float* zrow = xz + ((size_t)b * E2 + DI + d) * SEQ;
    float p3 = 0.f, p2 = 0.f, p1 = 0.f;               // rolling window
    for (int l = 0; l < SEQ; ++l) {
        int pl = (dir == 0) ? l : ((dir == 1) ? order[l] : order_rev[l]);
        float xv = xrow[pl];
        float c = fmaf(w0, p3, fmaf(w1, p2, fmaf(w2, p1, fmaf(w3, xv, bias))));
        float sc = c / (1.0f + __expf(-c));           // SiLU
        float zv = zrow[pl];
        float sz = zv / (1.0f + __expf(-zv));
        size_t o = ((size_t)s * SEQ + l) * DI + d;    // coalesced over d
        xct[o] = f2bf(sc);
        xcf[o] = sc;
        zs[o]  = sz;
        p3 = p2; p2 = p1; p1 = xv;
    }
}

// --------------------- chunked selective scan: pass 1 ----------------------
// Chunk c in [0, NCH-1): zero-seeded local scan; emit end-state h and decay
// product P per (s, c, n, d).  B matrix staged through LDS per 32 steps.
__global__ __launch_bounds__(256) void scan_pass1_kernel(
        const float* __restrict__ xdbl,   // (NS, 80, SEQ): rows 48..63 = B
        const float* __restrict__ dtl,    // (NS, SEQ, DI)
        const float* __restrict__ xcf,    // (NS, SEQ, DI)
        const float* __restrict__ A_log,  // (DI, DS)
        const float* __restrict__ bdt,    // (DI)
        float* __restrict__ hch,          // (NS, NCH-1, DS, DI)
        float* __restrict__ Pch) {        // (NS, NCH-1, DS, DI)
    __shared__ float Bs[DS][32];
    const int s = blockIdx.y;
    const int c = blockIdx.z;                        // 0 .. NCH-2
    const int d = blockIdx.x * 256 + threadIdx.x;

    float Ac[DS];
    #pragma unroll
    for (int n = 0; n < DS; ++n) Ac[n] = -__expf(A_log[d * DS + n]);
    const float bd = bdt[d];
    float h[DS], P[DS];
    #pragma unroll
    for (int n = 0; n < DS; ++n) { h[n] = 0.0f; P[n] = 1.0f; }

    const float* xd = xdbl + (size_t)s * 80 * SEQ;
    const int lbeg = c * CL;
    for (int l0 = lbeg; l0 < lbeg + CL; l0 += 32) {
        __syncthreads();
        for (int i = threadIdx.x; i < DS * 32; i += 256) {
            int r = i >> 5, cc = i & 31;
            Bs[r][cc] = xd[(48 + r) * SEQ + l0 + cc];
        }
        __syncthreads();
        for (int lc = 0; lc < 32; ++lc) {
            size_t o = ((size_t)s * SEQ + (l0 + lc)) * DI + d;
            float dt  = softplusf(dtl[o] + bd);
            float dtx = dt * xcf[o];
            #pragma unroll
            for (int n = 0; n < DS; ++n) {
                float dA = __expf(dt * Ac[n]);
                h[n] = fmaf(dA, h[n], dtx * Bs[n][lc]);
                P[n] *= dA;
            }
        }
    }
    #pragma unroll
    for (int n = 0; n < DS; ++n) {
        size_t oo = (((size_t)s * (NCH - 1) + c) * DS + n) * DI + d;
        hch[oo] = h[n];
        Pch[oo] = P[n];
    }
}

// ------------------- chunk-summary fix-up (serial over c) ------------------
// H0[s][0] = 0;  H0[s][c+1] = P[s][c] * H0[s][c] + h[s][c]
__global__ __launch_bounds__(256) void scan_fix_kernel(
        const float* __restrict__ hch, const float* __restrict__ Pch,
        float* __restrict__ H0) {       // (NS, NCH, DS, DI)
    int t = blockIdx.x * 256 + threadIdx.x;          // over NS*DS*DI
    if (t >= NS * DS * DI) return;
    int d = t % DI;
    int n = (t / DI) % DS;
    int s = t / (DI * DS);
    float h = 0.0f;
    H0[(((size_t)s * NCH + 0) * DS + n) * DI + d] = 0.0f;
    for (int c = 0; c < NCH - 1; ++c) {
        size_t oo = (((size_t)s * (NCH - 1) + c) * DS + n) * DI + d;
        h = fmaf(Pch[oo], h, hch[oo]);
        H0[(((size_t)s * NCH + c + 1) * DS + n) * DI + d] = h;
    }
}

// --------------------- chunked selective scan: pass 2 ----------------------
// Re-run chunk c seeded with H0, emit gated outputs ys.
__global__ __launch_bounds__(256) void scan_pass2_kernel(
        const float* __restrict__ xdbl,   // rows 48..79 = B, C
        const float* __restrict__ dtl,
        const float* __restrict__ xcf,
        const float* __restrict__ zs,     // silu(z)
        const float* __restrict__ A_log,
        const float* __restrict__ Dprm,
        const float* __restrict__ bdt,
        const float* __restrict__ H0,     // (NS, NCH, DS, DI)
        float* __restrict__ ys) {         // (NS, SEQ, DI)
    __shared__ float Bs[DS][32];
    __shared__ float Cs[DS][32];
    const int s = blockIdx.y;
    const int c = blockIdx.z;                        // 0 .. NCH-1
    const int d = blockIdx.x * 256 + threadIdx.x;

    float Ac[DS];
    #pragma unroll
    for (int n = 0; n < DS; ++n) Ac[n] = -__expf(A_log[d * DS + n]);
    const float Dv = Dprm[d];
    const float bd = bdt[d];
    float h[DS];
    #pragma unroll
    for (int n = 0; n < DS; ++n)
        h[n] = H0[(((size_t)s * NCH + c) * DS + n) * DI + d];

    const float* xd = xdbl + (size_t)s * 80 * SEQ;
    const int lbeg = c * CL;
    for (int l0 = lbeg; l0 < lbeg + CL; l0 += 32) {
        __syncthreads();
        for (int i = threadIdx.x; i < 32 * 32; i += 256) {
            int r = i >> 5, cc = i & 31;
            float v = xd[(48 + r) * SEQ + l0 + cc];
            if (r < DS) Bs[r][cc] = v; else Cs[r - DS][cc] = v;
        }
        __syncthreads();
        for (int lc = 0; lc < 32; ++lc) {
            size_t o = ((size_t)s * SEQ + (l0 + lc)) * DI + d;
            float dt  = softplusf(dtl[o] + bd);
            float xcv = xcf[o];
            float dtx = dt * xcv;
            float y = 0.0f;
            #pragma unroll
            for (int n = 0; n < DS; ++n) {
                float dA = __expf(dt * Ac[n]);
                h[n] = fmaf(dA, h[n], dtx * Bs[n][lc]);
                y = fmaf(h[n], Cs[n][lc], y);
            }
            ys[o] = (y + xcv * Dv) * zs[o];
        }
    }
}

// ------------------------ merge 3 directions -> bf16 -----------------------
__global__ __launch_bounds__(256) void merge_kernel(
        const float* __restrict__ ys, const int* __restrict__ inv_order,
        const int* __restrict__ inv_order_rev, bf16_t* __restrict__ ym) {
    int idx = blockIdx.x * 256 + threadIdx.x;        // over B_SZ*SEQ*DI
    if (idx >= B_SZ * SEQ * DI) return;
    int d = idx % DI;
    int l = (idx / DI) % SEQ;
    int b = idx / (DI * SEQ);
    float v = ys[(((size_t)(3 * b + 0) * SEQ) + l)                 * DI + d]
            + ys[(((size_t)(3 * b + 1) * SEQ) + inv_order[l])      * DI + d]
            + ys[(((size_t)(3 * b + 2) * SEQ) + inv_order_rev[l])  * DI + d];
    ym[idx] = f2bf(v);
}

// --------------------------------- launch ----------------------------------
extern "C" void kernel_launch(void* const* d_in, const int* in_sizes, int n_in,
                              void* d_out, int out_size, void* d_ws, size_t ws_size,
                              hipStream_t stream) {
    const float* x      = (const float*)d_in[0];
    const float* W_in   = (const float*)d_in[1];
    const float* conv_w = (const float*)d_in[2];
    const float* conv_b = (const float*)d_in[3];
    const float* W_xprj = (const float*)d_in[4];
    const float* W_dt   = (const float*)d_in[5];
    const float* b_dt   = (const float*)d_in[6];
    const float* A_log  = (const float*)d_in[7];
    const float* Dprm   = (const float*)d_in[8];
    const float* W_out  = (const float*)d_in[9];
    const int* order    = (const int*)d_in[10];
    const int* order_rev= (const int*)d_in[11];
    const int* inv_ord  = (const int*)d_in[12];
    const int* inv_ordr = (const int*)d_in[13];
    float* out = (float*)d_out;

    // ---- workspace carve-out (256B aligned) ----
    char* ws = (char*)d_ws;
    size_t off = 0;
    auto take = [&](size_t bytes) { char* p = ws + off;
                                    off = (off + bytes + 255) & ~(size_t)255;
                                    return p; };
    bf16_t* winb  = (bf16_t*)take((size_t)E2 * DM * 2);
    bf16_t* xb    = (bf16_t*)take((size_t)B_SZ * SEQ * DM * 2);
    bf16_t* wxpb  = (bf16_t*)take((size_t)80 * DI * 2);
    bf16_t* wdtb  = (bf16_t*)take((size_t)DI * RKP * 2);
    bf16_t* woutb = (bf16_t*)take((size_t)DM * DI * 2);
    float*  xzf   = (float*) take((size_t)B_SZ * E2 * SEQ * 4);
    bf16_t* xct   = (bf16_t*)take((size_t)NS * SEQ * DI * 2);
    float*  xcf   = (float*) take((size_t)NS * SEQ * DI * 4);
    float*  zsf   = (float*) take((size_t)NS * SEQ * DI * 4);
    float*  xdbl  = (float*) take((size_t)NS * 80 * SEQ * 4);
    bf16_t* dtrt  = (bf16_t*)take((size_t)NS * SEQ * RKP * 2);
    float*  dtlf  = (float*) take((size_t)NS * SEQ * DI * 4);
    float*  ysf   = (float*) take((size_t)NS * SEQ * DI * 4);
    bf16_t* ymb   = (bf16_t*)take((size_t)B_SZ * SEQ * DI * 2);
    float*  hch   = (float*) take((size_t)NS * (NCH - 1) * DS * DI * 4);
    float*  Pch   = (float*) take((size_t)NS * (NCH - 1) * DS * DI * 4);
    float*  H0    = (float*) take((size_t)NS * NCH * DS * DI * 4);
    (void)ws_size; (void)in_sizes; (void)n_in; (void)out_size;

    auto cvt = [&](const float* src, bf16_t* dst, int n) {
        cvt_bf16_kernel<<<(n + 255) / 256, 256, 0, stream>>>(src, dst, n);
    };

    // 1) precision conversion of GEMM operands
    cvt(W_in,   winb,  E2 * DM);
    cvt(x,      xb,    B_SZ * SEQ * DM);
    cvt(W_xprj, wxpb,  80 * DI);
    cvt(W_out,  woutb, DM * DI);
    cvt_wdt_kernel<<<(DI * RKP + 255) / 256, 256, 0, stream>>>(W_dt, wdtb);

    // 2) xz[b] (E2 x SEQ) = W_in (E2 x DM) * x[b]^T   (BT = x[b], natural)
    gemm_bf16_wmma<<<dim3(SEQ / 64, E2 / 64, B_SZ), 128, 0, stream>>>(
        winb, DM, 0, xb, DM, (long long)SEQ * DM,
        xzf, SEQ, (long long)E2 * SEQ, E2, SEQ, DM);

    // 3) depthwise conv + SiLU + permuted gather + SiLU(z)
    conv_silu_kernel<<<dim3(DI / 256, NS), 256, 0, stream>>>(
        xzf, conv_w, conv_b, order, order_rev, xct, xcf, zsf);

    // 4) x_dbl[s] (80 x SEQ) = W_xproj (80 x DI) * xc[s]   (BT = xct[s])
    gemm_bf16_wmma<<<dim3(SEQ / 64, (80 + 63) / 64, NS), 128, 0, stream>>>(
        wxpb, DI, 0, xct, DI, (long long)SEQ * DI,
        xdbl, SEQ, (long long)80 * SEQ, 80, SEQ, DI);

    // 5) repack dt_r -> (s, L, 64) bf16
    cvt_dtr_kernel<<<(NS * SEQ + 255) / 256, 256, 0, stream>>>(xdbl, dtrt);

    // 6) dt_lin[s] (SEQ x DI) = dt_r^T (SEQ x 64) * W_dt^T  (BT = W_dt padded)
    gemm_bf16_wmma<<<dim3(DI / 64, SEQ / 64, NS), 128, 0, stream>>>(
        dtrt, RKP, (long long)SEQ * RKP, wdtb, RKP, 0,
        dtlf, DI, (long long)SEQ * DI, SEQ, DI, RKP);

    // 7) chunk-parallel selective scan
    scan_pass1_kernel<<<dim3(DI / 256, NS, NCH - 1), 256, 0, stream>>>(
        xdbl, dtlf, xcf, A_log, b_dt, hch, Pch);
    scan_fix_kernel<<<(NS * DS * DI + 255) / 256, 256, 0, stream>>>(
        hch, Pch, H0);
    scan_pass2_kernel<<<dim3(DI / 256, NS, NCH), 256, 0, stream>>>(
        xdbl, dtlf, xcf, zsf, A_log, Dprm, b_dt, H0, ysf);

    // 8) merge the 3 directions with inverse permutations -> bf16
    merge_kernel<<<(B_SZ * SEQ * DI + 255) / 256, 256, 0, stream>>>(
        ysf, inv_ord, inv_ordr, ymb);

    // 9) out (B*L x DM) = y_merged (B*L x DI) * W_out^T  (BT = W_out, natural)
    gemm_bf16_wmma<<<dim3(DM / 64, (B_SZ * SEQ) / 64, 1), 128, 0, stream>>>(
        ymb, DI, 0, woutb, DI, 0,
        out, DM, 0, B_SZ * SEQ, DM, DI);
}
